// GlobalSelfFeastBlock_66133906424007
// MI455X (gfx1250) — matrix operand order repaired and checked by reading, not compiled, not run on hardware
//
// GlobalSelfFeastBlock for MI455X (gfx1250): WMMA f16 GEMMs + dual-softmax
// flash attention with double-buffered async-to-LDS K/V staging.
// f32 accumulation, f32 softmax/LN math, f16 matrix inputs.
#include <hip/hip_runtime.h>

typedef _Float16 half_t;
typedef __attribute__((ext_vector_type(16))) _Float16 v16h;
typedef __attribute__((ext_vector_type(8)))  _Float16 v8h;
typedef __attribute__((ext_vector_type(8)))  float    v8f;

#define DEV __device__ __forceinline__

// ---------------------------------------------------------------------------
// WMMA helper: D = A(16x32 f16) * B(32x16 f16) + C(16x16 f32)
DEV v8f wmma16x16x32(v16h a, v16h b, v8f c) {
  return __builtin_amdgcn_wmma_f32_16x16x32_f16(
      /*neg_a=*/false, a, /*neg_b=*/false, b,
      /*c_mod=*/(short)0, c, /*reuse_a=*/false, /*reuse_b=*/false);
}

// A-fragment (16x32, row-major source): lane l holds row m0+(l&15),
// elems[0..15] = K = k0 + (l>>4)*16 .. +15.
DEV v16h load_afrag(const half_t* __restrict__ A, int lda, int m0, int k0) {
  const int lane = threadIdx.x & 31;
  const half_t* p = A + (size_t)(m0 + (lane & 15)) * lda + (k0 + ((lane >> 4) << 4));
  v8h lo = *(const v8h*)p;
  v8h hi = *(const v8h*)(p + 8);
  v16h r;
#pragma unroll
  for (int i = 0; i < 8; ++i) { r[i] = lo[i]; r[i + 8] = hi[i]; }
  return r;
}

// B-fragment for C = A * W^T : B[k][n] = W[n][k] (W row-major, ld = K).
// lane l holds col n0+(l&15), elems[0..15] = K = k0 + (l>>4)*16 .. +15.
DEV v16h load_bfrag(const half_t* __restrict__ W, int ldw, int n0, int k0) {
  const int lane = threadIdx.x & 31;
  const half_t* p = W + (size_t)(n0 + (lane & 15)) * ldw + (k0 + ((lane >> 4) << 4));
  return *(const v16h*)p;  // 32B, split by compiler into 2x b128
}

// XOR-shuffle via LDS swizzle hardware (group-of-32 mode, no LDS memory use).
template <int XM>
DEV float swz_xor_f(float v) {
  int i = __builtin_amdgcn_ds_swizzle(__float_as_int(v), (XM << 10) | 0x1f);
  return __int_as_float(i);
}

// Async DMA: 16B global -> LDS, tracked by ASYNCcnt (gfx1250).
// LDS address = low 32 bits of the generic pointer (aperture rule: addr[31:0]).
DEV void async_ld16(void* lds_dst, const half_t* g) {
  const unsigned int lo = (unsigned int)(uintptr_t)lds_dst;
  asm volatile("global_load_async_to_lds_b128 %0, %1, off"
               :: "v"(lo), "v"((unsigned long long)(uintptr_t)g)
               : "memory");
}
DEV void wait_asynccnt0() {
  asm volatile("s_wait_asynccnt 0x0" ::: "memory");
}

// ---------------------------------------------------------------------------
__global__ void f32_to_f16(const float* __restrict__ in, half_t* __restrict__ out, int n) {
  int i = blockIdx.x * blockDim.x + threadIdx.x;
  if (i < n) out[i] = (half_t)in[i];
}

// ---------------------------------------------------------------------------
// GEMM: C[M,Nout] = A[M,K] @ W[Nout,K]^T, per-wave 16x64 tile.
enum { EPI_HALF = 0, EPI_HALF_T = 1, EPI_F32_BIAS_RES = 2, EPI_HALF_BIAS_GELU = 3 };

template <int EPI>
__global__ __launch_bounds__(256) void gemm_xWT(
    const half_t* __restrict__ A, const half_t* __restrict__ W,
    int M, int Nout, int K,
    float* __restrict__ outF, half_t* __restrict__ outH,
    const float* __restrict__ bias, const float* __restrict__ res) {
  const int wid = blockIdx.x * 8 + (threadIdx.x >> 5);
  const int nT  = Nout >> 6;
  const int mt  = wid / nT, nt = wid % nT;
  if (mt >= (M >> 4)) return;
  const int m0 = mt << 4, n0 = nt << 6;
  const int lane = threadIdx.x & 31;

  v8f acc[4] = {};
  for (int k0 = 0; k0 < K; k0 += 32) {
    // Hoist all fragment loads ahead of the wmma chain so the loads issue as
    // one clause and the waits can drain progressively.
    const v16h a = load_afrag(A, K, m0, k0);
    v16h b[4];
#pragma unroll
    for (int n = 0; n < 4; ++n) b[n] = load_bfrag(W, K, n0 + n * 16, k0);
    if (k0 + 32 < K) {  // warm next K-step lines (global_prefetch_b8)
      __builtin_prefetch(A + (size_t)(m0 + (lane & 15)) * K + k0 + 32, 0, 3);
      __builtin_prefetch(W + (size_t)(n0 + (lane & 15)) * K + k0 + 32, 0, 3);
    }
#pragma unroll
    for (int n = 0; n < 4; ++n) acc[n] = wmma16x16x32(a, b[n], acc[n]);
  }

  const int hs = lane >> 4, cl = lane & 15;
#pragma unroll
  for (int n = 0; n < 4; ++n) {
    const int col = n0 + n * 16 + cl;
#pragma unroll
    for (int j = 0; j < 8; ++j) {
      const int row = m0 + j + 8 * hs;
      const float v = acc[n][j];
      if constexpr (EPI == EPI_HALF) {
        outH[(size_t)row * Nout + col] = (half_t)v;
      } else if constexpr (EPI == EPI_HALF_T) {
        outH[(size_t)col * M + row] = (half_t)v;   // transposed (V^T per head)
      } else if constexpr (EPI == EPI_F32_BIAS_RES) {
        const size_t idx = (size_t)row * Nout + col;
        outF[idx] = v + bias[col] + res[idx];
      } else {  // bias + exact-erf GELU -> f16
        const float xg = v + bias[col];
        outH[(size_t)row * Nout + col] =
            (half_t)(0.5f * xg * (1.0f + erff(xg * 0.70710678118654752f)));
      }
    }
  }
}

// ---------------------------------------------------------------------------
// Flash attention with two online softmaxes (pos & neg share S and V tiles).
// Block = 8 waves, one head, 8 consecutive 16-row query tiles. All waves march
// over the same key blocks, so K (32x128) and V^T (128x32) tiles are staged in
// LDS once per block via async-to-LDS DMA, double buffered.
__global__ __launch_bounds__(256) void attn_kernel(
    const half_t* __restrict__ Qh, const half_t* __restrict__ Kh,
    const half_t* __restrict__ Vt,  // [D][N]: Vt[h*HD+d][m] = V[m][h*HD+d]
    const int* __restrict__ coords, half_t* __restrict__ attOut) {
  constexpr int N = 4096, D = 1024, HD = 128;
  __shared__ __align__(32) _Float16 kbuf[2][32][128];  // [key][d-slice]
  __shared__ __align__(32) _Float16 vbuf[2][128][32];  // [d-slice][key]
  __shared__ __align__(32) _Float16 plds[8][16 * 32];

  const int wv = threadIdx.x >> 5;
  const int lane = threadIdx.x & 31;
  const int hsel = lane >> 4, cl = lane & 15;
  const int head = blockIdx.y;
  const int q0 = (blockIdx.x * 8 + wv) << 4;
  const int hc = head * HD;
  const float hscale = exp2f(-(float)(head + 1));
  constexpr float inv_sqrt_hd = 0.08838834764831845f;  // 1/sqrt(128)

  // Stage one K/V key-block (32 keys) into LDS buffer ib: 512 16B chunks each,
  // 2 per thread.
  auto stage_kv = [&](int ib, int j0) {
    const int tid = threadIdx.x;
#pragma unroll
    for (int t = 0; t < 2; ++t) {
      const int c = tid + t * 256;
      const int kr = c >> 4, kc = (c & 15) << 3;            // 32 x 16 chunks
      async_ld16(&kbuf[ib][kr][kc], Kh + (size_t)(j0 + kr) * D + hc + kc);
      const int vr = c >> 2, vc = (c & 3) << 3;             // 128 x 4 chunks
      async_ld16(&vbuf[ib][vr][vc], Vt + (size_t)(hc + vr) * N + j0 + vc);
    }
  };

  v16h aq[4];
#pragma unroll
  for (int k = 0; k < 4; ++k) aq[k] = load_afrag(Qh, D, q0, hc + k * 32);

  float qx[8], qy[8];
#pragma unroll
  for (int r = 0; r < 8; ++r) {
    const int row = q0 + r + 8 * hsel;
    qx[r] = (float)coords[row * 2 + 0];
    qy[r] = (float)coords[row * 2 + 1];
  }

  float m_p[8], l_p[8], m_n[8], l_n[8];
#pragma unroll
  for (int r = 0; r < 8; ++r) { m_p[r] = m_n[r] = -1e30f; l_p[r] = l_n[r] = 0.f; }
  v8f o_p[8] = {}, o_n[8] = {};

  stage_kv(0, 0);  // prologue: fill buffer 0

  for (int it = 0; it < N / 32; ++it) {
    const int j0 = it * 32;
    const int ib = it & 1;
    wait_asynccnt0();     // this thread's async copies for buffer ib are in LDS
    __syncthreads();      // every thread's copies are in LDS
    if (j0 + 32 < N) stage_kv(ib ^ 1, j0 + 32);  // prefetch next block

    // S = Q K^T for 16 rows x 32 keys (two 16x16 subtiles), B-frags from LDS.
    v8f s[2] = {};
#pragma unroll
    for (int t = 0; t < 2; ++t)
#pragma unroll
      for (int k = 0; k < 4; ++k) {
        const v16h b = *(const v16h*)&kbuf[ib][t * 16 + cl][k * 32 + hsel * 16];
        s[t] = wmma16x16x32(aq[k], b, s[t]);
      }

    float kx[2], ky[2];
#pragma unroll
    for (int t = 0; t < 2; ++t) {
      const int c = j0 + t * 16 + cl;
      kx[t] = (float)coords[c * 2 + 0];
      ky[t] = (float)coords[c * 2 + 1];
    }

    float lp[2][8], lng[2][8], al_p[8], al_n[8];
#pragma unroll
    for (int t = 0; t < 2; ++t)
#pragma unroll
      for (int r = 0; r < 8; ++r) {
        const float sc = s[t][r] * inv_sqrt_hd;
        const float dx = qx[r] - kx[t], dy = qy[r] - ky[t];
        const float bias = sqrtf(dx * dx + dy * dy) * hscale;
        lp[t][r]  = sc - bias;                       // pos logits
        lng[t][r] = (-sc - bias) * (1.0f / 0.6f);    // neg logits / tau
      }

    // Dual online softmax. Row r lives in one 16-lane half; XOR 1/2/4/8
    // reduces across the 16 columns held by that half.
#pragma unroll
    for (int r = 0; r < 8; ++r) {
      {
        float mx = fmaxf(lp[0][r], lp[1][r]);
        mx = fmaxf(mx, swz_xor_f<1>(mx)); mx = fmaxf(mx, swz_xor_f<2>(mx));
        mx = fmaxf(mx, swz_xor_f<4>(mx)); mx = fmaxf(mx, swz_xor_f<8>(mx));
        const float nm = fmaxf(m_p[r], mx);
        const float al = __expf(m_p[r] - nm);
        const float e0 = __expf(lp[0][r] - nm), e1 = __expf(lp[1][r] - nm);
        lp[0][r] = e0; lp[1][r] = e1;
        float sm = e0 + e1;
        sm += swz_xor_f<1>(sm); sm += swz_xor_f<2>(sm);
        sm += swz_xor_f<4>(sm); sm += swz_xor_f<8>(sm);
        l_p[r] = l_p[r] * al + sm; m_p[r] = nm; al_p[r] = al;
      }
      {
        float mx = fmaxf(lng[0][r], lng[1][r]);
        mx = fmaxf(mx, swz_xor_f<1>(mx)); mx = fmaxf(mx, swz_xor_f<2>(mx));
        mx = fmaxf(mx, swz_xor_f<4>(mx)); mx = fmaxf(mx, swz_xor_f<8>(mx));
        const float nm = fmaxf(m_n[r], mx);
        const float al = __expf(m_n[r] - nm);
        const float e0 = __expf(lng[0][r] - nm), e1 = __expf(lng[1][r] - nm);
        lng[0][r] = e0; lng[1][r] = e1;
        float sm = e0 + e1;
        sm += swz_xor_f<1>(sm); sm += swz_xor_f<2>(sm);
        sm += swz_xor_f<4>(sm); sm += swz_xor_f<8>(sm);
        l_n[r] = l_n[r] * al + sm; m_n[r] = nm; al_n[r] = al;
      }
    }

    // Rescale output accumulators (C-layout: VGPR j <-> row j / j+8).
#pragma unroll
    for (int dt = 0; dt < 8; ++dt)
#pragma unroll
      for (int j = 0; j < 8; ++j) { o_p[dt][j] *= al_p[j]; o_n[dt][j] *= al_n[j]; }

    // Re-layout P (C-layout -> A-fragment) through this wave's LDS tile.
    _Float16* pb = plds[wv];
#pragma unroll
    for (int t = 0; t < 2; ++t)
#pragma unroll
      for (int r = 0; r < 8; ++r)
        pb[(r + 8 * hsel) * 32 + t * 16 + cl] = (half_t)lp[t][r];
    __builtin_amdgcn_wave_barrier();
    v16h pfp;
    {
      const _Float16* pp = pb + cl * 32 + hsel * 16;
      v8h lo = *(const v8h*)pp;
      v8h hi = *(const v8h*)(pp + 8);
#pragma unroll
      for (int i = 0; i < 8; ++i) { pfp[i] = lo[i]; pfp[i + 8] = hi[i]; }
    }
    __builtin_amdgcn_wave_barrier();
#pragma unroll
    for (int t = 0; t < 2; ++t)
#pragma unroll
      for (int r = 0; r < 8; ++r)
        pb[(r + 8 * hsel) * 32 + t * 16 + cl] = (half_t)lng[t][r];
    __builtin_amdgcn_wave_barrier();
    v16h pfn;
    {
      const _Float16* pp = pb + cl * 32 + hsel * 16;
      v8h lo = *(const v8h*)pp;
      v8h hi = *(const v8h*)(pp + 8);
#pragma unroll
      for (int i = 0; i < 8; ++i) { pfn[i] = lo[i]; pfn[i + 8] = hi[i]; }
    }
    __builtin_amdgcn_wave_barrier();

    // O += P @ V ; V^T fragments (LDS) shared by pos and neg accumulators.
#pragma unroll
    for (int dt = 0; dt < 8; ++dt) {
      const v16h bv = *(const v16h*)&vbuf[ib][dt * 16 + cl][hsel * 16];
      o_p[dt] = wmma16x16x32(pfp, bv, o_p[dt]);
      o_n[dt] = wmma16x16x32(pfn, bv, o_n[dt]);
    }
  }

  float rp[8], rn[8];
#pragma unroll
  for (int r = 0; r < 8; ++r) { rp[r] = 1.0f / l_p[r]; rn[r] = 1.5f / l_n[r]; }
#pragma unroll
  for (int dt = 0; dt < 8; ++dt)
#pragma unroll
    for (int j = 0; j < 8; ++j) {
      const float v = o_p[dt][j] * rp[j] - o_n[dt][j] * rn[j];
      attOut[(size_t)(q0 + j + 8 * hsel) * D + hc + dt * 16 + cl] = (half_t)v;
    }
}

// ---------------------------------------------------------------------------
// Row LayerNorm: one block per row, f32 math, optional f16 mirror.
__global__ __launch_bounds__(256) void layernorm_kernel(
    const float* __restrict__ x, const float* __restrict__ g,
    const float* __restrict__ bta, float* __restrict__ outF,
    half_t* __restrict__ outH, int D) {
  __shared__ float redS[8], redS2[8];
  const int row = blockIdx.x;
  const float* xr = x + (size_t)row * D;
  float s = 0.f, s2 = 0.f;
  for (int c = threadIdx.x; c < D; c += blockDim.x) { const float v = xr[c]; s += v; s2 += v * v; }
  s  += swz_xor_f<1>(s);  s  += swz_xor_f<2>(s);  s  += swz_xor_f<4>(s);
  s  += swz_xor_f<8>(s);  s  += swz_xor_f<16>(s);
  s2 += swz_xor_f<1>(s2); s2 += swz_xor_f<2>(s2); s2 += swz_xor_f<4>(s2);
  s2 += swz_xor_f<8>(s2); s2 += swz_xor_f<16>(s2);
  const int wv = threadIdx.x >> 5, lane = threadIdx.x & 31;
  if (lane == 0) { redS[wv] = s; redS2[wv] = s2; }
  __syncthreads();
  float tot = 0.f, tot2 = 0.f;
#pragma unroll
  for (int i = 0; i < 8; ++i) { tot += redS[i]; tot2 += redS2[i]; }
  const float mu = tot / (float)D;
  const float var = tot2 / (float)D - mu * mu;
  const float rstd = rsqrtf(var + 1e-5f);
  for (int c = threadIdx.x; c < D; c += blockDim.x) {
    const float y = (xr[c] - mu) * rstd * g[c] + bta[c];
    outF[(size_t)row * D + c] = y;
    if (outH) outH[(size_t)row * D + c] = (half_t)y;
  }
}

// ---------------------------------------------------------------------------
extern "C" void kernel_launch(void* const* d_in, const int* in_sizes, int n_in,
                              void* d_out, int out_size, void* d_ws, size_t ws_size,
                              hipStream_t stream) {
  (void)in_sizes; (void)n_in; (void)out_size; (void)ws_size;
  const int N = 4096, D = 1024, FFN = 1536;

  const float* features = (const float*)d_in[0];
  const int*   coords   = (const int*)d_in[1];
  const float* Wq = (const float*)d_in[2];
  const float* Wk = (const float*)d_in[3];
  const float* Wv = (const float*)d_in[4];
  const float* Wo = (const float*)d_in[5];
  const float* bo = (const float*)d_in[6];
  const float* ln1_g = (const float*)d_in[7];
  const float* ln1_b = (const float*)d_in[8];
  const float* W1 = (const float*)d_in[9];
  const float* b1 = (const float*)d_in[10];
  const float* W2 = (const float*)d_in[11];
  const float* b2 = (const float*)d_in[12];
  const float* ln2_g = (const float*)d_in[13];
  const float* ln2_b = (const float*)d_in[14];
  float* out = (float*)d_out;

  char* base = (char*)d_ws;
  size_t off = 0;
  auto alloc = [&](size_t bytes) -> void* {
    void* p = base + off;
    off = (off + bytes + 255) & ~(size_t)255;
    return p;
  };
  half_t* feat_h = (half_t*)alloc((size_t)N * D * 2);
  half_t* wq_h = (half_t*)alloc((size_t)D * D * 2);
  half_t* wk_h = (half_t*)alloc((size_t)D * D * 2);
  half_t* wv_h = (half_t*)alloc((size_t)D * D * 2);
  half_t* wo_h = (half_t*)alloc((size_t)D * D * 2);
  half_t* w1_h = (half_t*)alloc((size_t)FFN * D * 2);
  half_t* w2_h = (half_t*)alloc((size_t)D * FFN * 2);
  half_t* q_h  = (half_t*)alloc((size_t)N * D * 2);   // [N,D]
  half_t* k_h  = (half_t*)alloc((size_t)N * D * 2);   // [N,D]
  half_t* vt_h = (half_t*)alloc((size_t)N * D * 2);   // [D,N] transposed
  half_t* att_h = (half_t*)alloc((size_t)N * D * 2);  // [N,D]
  float*  x0 = (float*)alloc((size_t)N * D * 4);      // pre-LN1 (reused as y2)
  float*  x1 = (float*)alloc((size_t)N * D * 4);      // post-LN1
  half_t* x1_h = (half_t*)alloc((size_t)N * D * 2);
  half_t* h_h = q_h;  // reuse Q/K region (dead after attention) for FFN hidden
  float*  y2 = x0;    // reuse pre-LN1 buffer (dead after LN1)

  auto cvt = [&](const float* src, half_t* dst, int n) {
    f32_to_f16<<<dim3((n + 255) / 256), dim3(256), 0, stream>>>(src, dst, n);
  };
  cvt(features, feat_h, N * D);
  cvt(Wq, wq_h, D * D);
  cvt(Wk, wk_h, D * D);
  cvt(Wv, wv_h, D * D);
  cvt(Wo, wo_h, D * D);
  cvt(W1, w1_h, FFN * D);
  cvt(W2, w2_h, D * FFN);

  const int gemmBlk_D  = (N / 16) * (D / 64) / 8;    // 512
  const int gemmBlk_F  = (N / 16) * (FFN / 64) / 8;  // 768

  // QKV projections
  gemm_xWT<EPI_HALF><<<dim3(gemmBlk_D), dim3(256), 0, stream>>>(
      feat_h, wq_h, N, D, D, nullptr, q_h, nullptr, nullptr);
  gemm_xWT<EPI_HALF><<<dim3(gemmBlk_D), dim3(256), 0, stream>>>(
      feat_h, wk_h, N, D, D, nullptr, k_h, nullptr, nullptr);
  gemm_xWT<EPI_HALF_T><<<dim3(gemmBlk_D), dim3(256), 0, stream>>>(
      feat_h, wv_h, N, D, D, nullptr, vt_h, nullptr, nullptr);

  // Dual-softmax flash attention: grid = (N/16/8 qtile-blocks, H heads)
  attn_kernel<<<dim3(32, 8), dim3(256), 0, stream>>>(q_h, k_h, vt_h, coords, att_h);

  // Output projection + bias + residual, then LN1 (f32 + f16 mirror)
  gemm_xWT<EPI_F32_BIAS_RES><<<dim3(gemmBlk_D), dim3(256), 0, stream>>>(
      att_h, wo_h, N, D, D, x0, nullptr, bo, features);
  layernorm_kernel<<<dim3(N), dim3(256), 0, stream>>>(x0, ln1_g, ln1_b, x1, x1_h, D);

  // FFN: GELU(x1 W1^T + b1) -> f16, then W2^T + b2 + residual(x1), then LN2
  gemm_xWT<EPI_HALF_BIAS_GELU><<<dim3(gemmBlk_F), dim3(256), 0, stream>>>(
      x1_h, w1_h, N, FFN, D, nullptr, h_h, b1, nullptr);
  gemm_xWT<EPI_F32_BIAS_RES><<<dim3(gemmBlk_D), dim3(256), 0, stream>>>(
      h_h, w2_h, N, D, FFN, y2, nullptr, b2, x1);
  layernorm_kernel<<<dim3(N), dim3(256), 0, stream>>>(y2, ln2_g, ln2_b, out, nullptr, D);
}
